// VectorQuantizer_17188459119103
// MI455X (gfx1250) — matrix operand order repaired and checked by reading, not compiled, not run on hardware
//
#include <hip/hip_runtime.h>

// ---------------------------------------------------------------------------
// VQ-VAE vector quantizer for MI455X (gfx1250, wave32, WMMA).
//
// dist_k = ||z||^2 + ||e_k||^2 - 2 z.e_k ; argmin over k only needs
// ||e_k||^2 - 2 z.e_k. The 131072x1024x64 dot-product GEMM is done with
// v_wmma_f32_16x16x32_f16 (f16 in, f32 accum) -> memory-bound kernel
// (~96MB moved @ 23.3TB/s). Output z_q gathered from the ORIGINAL f32
// codebook, so forward values are full precision.
// ---------------------------------------------------------------------------

typedef _Float16 half16 __attribute__((ext_vector_type(16)));
typedef float    f8v    __attribute__((ext_vector_type(8)));
typedef float    f4v    __attribute__((ext_vector_type(4)));

#define NCODES   1024
#define CDIM     64
#define NROWS    131072          // 32*64*64 spatial vectors
#define ZQ_ELEMS 8388608         // 32*64*64*64
#define IDX_OFF  ZQ_ELEMS
#define LOSS_OFF (ZQ_ELEMS + NROWS)

#define CB_STRIDE_H 72           // LDS codebook row stride in halves (pad vs 64)
#define STG_STRIDE  68           // staging row stride in floats (pad vs 64)

// LDS layout (dynamic, 186880 bytes total):
//   [0,      147456)  f16 codebook, 1024 rows x 72 halves
//   [147456, 151552)  e2[1024] f32
//   [151552, 152064)  per-wave argmin indices, 8 waves x 16 ints
//   [152064, 186880)  per-wave f32 staging, 8 waves x 16 x 68 floats
#define SMEM_BYTES 186880

// ---------------- prep: codebook f32 -> f16, e2 = ||e||^2, zero loss --------
__global__ __launch_bounds__(256) void vq_prep(const float* __restrict__ cb,
                                               _Float16* __restrict__ cb16,
                                               float* __restrict__ e2,
                                               float* __restrict__ out) {
  int k = blockIdx.x * blockDim.x + threadIdx.x;
  if (k == 0 && blockIdx.x == 0) out[LOSS_OFF] = 0.0f;
  if (k < NCODES) {
    const float* row = cb + k * CDIM;
    _Float16* drow = cb16 + k * CDIM;
    float s = 0.0f;
#pragma unroll
    for (int d = 0; d < CDIM; ++d) {
      float v = row[d];
      s = __builtin_fmaf(v, v, s);
      drow[d] = (_Float16)v;
    }
    e2[k] = s;
  }
}

// ---------------- main: WMMA argmin + gather + loss -------------------------
__global__ __launch_bounds__(256) void vq_main(const float* __restrict__ z,
                                               const float* __restrict__ cb,
                                               const _Float16* __restrict__ cb16,
                                               const float* __restrict__ e2g,
                                               float* __restrict__ out) {
  extern __shared__ char smem[];
  _Float16* cbL  = (_Float16*)smem;
  float*    e2L  = (float*)(smem + 147456);
  int*      idxL = (int*)(smem + 151552);
  float*    stgL = (float*)(smem + 152064);

  const int tid = threadIdx.x;

  // Cooperative load: f16 codebook -> padded LDS (dword copies), e2 -> LDS.
  {
    const unsigned* src = (const unsigned*)cb16;
    unsigned* dst = (unsigned*)cbL;
    for (int g = tid; g < NCODES * CDIM / 2; g += 256) {
      int r = g >> 5, c = g & 31;          // 32 dwords per codebook row
      dst[r * 36 + c] = src[g];            // 36 dwords = 72 halves stride
    }
    for (int g = tid; g < NCODES; g += 256) e2L[g] = e2g[g];
  }
  __syncthreads();

  const int lane   = tid & 31;
  const int wave   = tid >> 5;
  const int lane16 = lane & 15;
  const int hi     = lane >> 4;
  const int rowBase = (blockIdx.x * 8 + wave) * 16;   // 16 z-rows per wave
  const int myRow   = rowBase + lane16;
  const int b   = myRow >> 12;                        // batch
  const int rem = myRow & 4095;                       // h*64 + w
  const float* zrow = z + (size_t)b * 262144 + rem;   // element d at zrow[d*4096]

  // A-operands (16x32 f16): lanes 0-15 row M=lane16 K={k0..k0+7, k0+16..k0+23},
  // lanes 16-31 same row K shifted by 8 (per ISA 16-bit A layout).
  half16 a0, a1;
#pragma unroll
  for (int j = 0; j < 8; ++j) {
    a0[j]     = (_Float16)zrow[( 0 + hi * 8 + j) * 4096];
    a0[j + 8] = (_Float16)zrow[(16 + hi * 8 + j) * 4096];
    a1[j]     = (_Float16)zrow[(32 + hi * 8 + j) * 4096];
    a1[j + 8] = (_Float16)zrow[(48 + hi * 8 + j) * 4096];
  }

  float minv[8];
  int   mini[8];
#pragma unroll
  for (int j = 0; j < 8; ++j) { minv[j] = 3.4e38f; mini[j] = 0; }

  union H16 { half16 v; f4v f[2]; };

  // Sweep all 1024 codes in 16-wide tiles; 2 chained WMMAs cover K=64.
#pragma unroll 2
  for (int t = 0; t < 64; ++t) {
    const int code = t * 16 + lane16;
    // B-operand (32x16 f16): lane = column N, K={k0+8h..+7, k0+16+8h..+7}
    // -> two 8-half chunks of a codebook row (matches A layout).
    const _Float16* brow = cbL + (size_t)code * CB_STRIDE_H + hi * 8;
    H16 b0, b1;
    b0.f[0] = *(const f4v*)(brow);
    b0.f[1] = *(const f4v*)(brow + 16);
    b1.f[0] = *(const f4v*)(brow + 32);
    b1.f[1] = *(const f4v*)(brow + 48);

    f8v acc = {0.f, 0.f, 0.f, 0.f, 0.f, 0.f, 0.f, 0.f};
    acc = __builtin_amdgcn_wmma_f32_16x16x32_f16(false, a0, false, b0.v,
                                                 (short)0, acc, false, false);
    acc = __builtin_amdgcn_wmma_f32_16x16x32_f16(false, a1, false, b1.v,
                                                 (short)0, acc, false, false);

    const float e2v = e2L[code];
    // C-layout: acc[j] = (row rowBase + hi*8 + j, column lane16) -> all 8
    // values of this lane share one candidate code.
#pragma unroll
    for (int j = 0; j < 8; ++j) {
      float dist = __builtin_fmaf(-2.0f, acc[j], e2v);
      bool lt = dist < minv[j];
      minv[j] = lt ? dist : minv[j];
      mini[j] = lt ? code : mini[j];
    }
  }

  // Cross-lane argmin within each 16-lane group (lowest-index tie-break to
  // match jnp.argmin semantics).
#pragma unroll
  for (int j = 0; j < 8; ++j) {
#pragma unroll
    for (int off = 8; off >= 1; off >>= 1) {
      float ov = __shfl_xor(minv[j], off, 16);
      int   oi = __shfl_xor(mini[j], off, 16);
      bool take = (ov < minv[j]) || (ov == minv[j] && oi < mini[j]);
      minv[j] = take ? ov : minv[j];
      mini[j] = take ? oi : mini[j];
    }
  }
  if (lane16 == 0) {
#pragma unroll
    for (int j = 0; j < 8; ++j) idxL[wave * 16 + hi * 8 + j] = mini[j];
  }
  __syncthreads();

  const int myIdx = idxL[wave * 16 + lane16];
  if (hi == 0) out[IDX_OFF + rowBase + lane16] = (float)myIdx;

  // Stage the selected f32 codebook rows into padded LDS so the [B,D,H,W]
  // stores below are coalesced across w.
  {
    const float* crow = cb + (size_t)myIdx * CDIM + hi * 32;
    float* srow = stgL + wave * (16 * STG_STRIDE) + lane16 * STG_STRIDE + hi * 32;
#pragma unroll
    for (int q = 0; q < 8; ++q)
      *(f4v*)(srow + q * 4) = *(const f4v*)(crow + q * 4);
  }
  __syncthreads();

  // Write z_q_st (== z_q forward value) and accumulate the fused VQ loss:
  // vq_loss = (1 + 0.25) * mean((z_q - z_e)^2).
  const float* stg = stgL + wave * (16 * STG_STRIDE);
  float* orow = out + (size_t)b * 262144 + rem;
  float lsum = 0.0f;
#pragma unroll 8
  for (int dd = 0; dd < 32; ++dd) {
    int d = hi * 32 + dd;
    float qv = stg[lane16 * STG_STRIDE + d];
    float zv = zrow[d * 4096];
    orow[d * 4096] = qv;
    float df = qv - zv;
    lsum = __builtin_fmaf(df, df, lsum);
  }
#pragma unroll
  for (int off = 16; off >= 1; off >>= 1) lsum += __shfl_xor(lsum, off, 32);
  if (lane == 0)
    atomicAdd(out + LOSS_OFF, lsum * (1.25f / 8388608.0f));
}

// ---------------------------------------------------------------------------
extern "C" void kernel_launch(void* const* d_in, const int* in_sizes, int n_in,
                              void* d_out, int out_size, void* d_ws, size_t ws_size,
                              hipStream_t stream) {
  (void)in_sizes; (void)n_in; (void)out_size; (void)ws_size;
  const float* z_e      = (const float*)d_in[0];   // [32,64,64,64] f32
  const float* codebook = (const float*)d_in[1];   // [1024,64] f32
  float* out = (float*)d_out;                      // z_q | indices(float) | loss

  _Float16* cb16 = (_Float16*)d_ws;                       // 128 KB
  float*    e2   = (float*)((char*)d_ws + 131072);        // 4 KB

  // Allow >64KB dynamic LDS (gfx1250 WGP has 320KB). Idempotent, capture-safe.
  (void)hipFuncSetAttribute((const void*)vq_main,
                            hipFuncAttributeMaxDynamicSharedMemorySize,
                            SMEM_BYTES);

  vq_prep<<<NCODES / 256, 256, 0, stream>>>(codebook, cb16, e2, out);
  vq_main<<<NROWS / 128, 256, SMEM_BYTES, stream>>>(z_e, codebook, cb16, e2, out);
}
